// EIRNN_33337536151745
// MI455X (gfx1250) — compile-verified, matrix operand order checked by reading, not run on hardware
//
#include <hip/hip_runtime.h>
#include <hip/hip_bf16.h>
#include <stdint.h>

// ---------------- problem constants ----------------
#define T_STEPS 256
#define B_SZ    256
#define IN_SZ   160     // 64 heading + 64 targcolor + 32 rules
#define IN_HT   128     // IN_H + IN_T
#define IN_Hh   64
#define H_SZ    1792
#define H1_SZ   512
#define H2_SZ   1024
#define H12_SZ  1536    // H1 + H2
#define ALPHA_F 0.333333333333f   // DT/TAU = 10/30
#define SIG_F   0.1f

#define NTILE_BLK 7     // H / 256 column groups
#define MTILE_BLK 16    // B / 16  row groups
#define NBLOCKS   (NTILE_BLK * MTILE_BLK)   // 112 persistent blocks
#define THREADS   128                        // 4 waves; each wave: 16M x 64N

// LDS A tile (padded +8 bf16/row: stride 900 dwords == 4 mod 64 -> conflict-free)
#define LDA  (H_SZ + 8)    // 1800

#define GLOBAL_AS __attribute__((address_space(1)))
#define LOCAL_AS  __attribute__((address_space(3)))

typedef __attribute__((ext_vector_type(16))) __bf16 bf16x16;
typedef __attribute__((ext_vector_type(8)))  __bf16 bf16x8;
typedef __attribute__((ext_vector_type(8)))  float  f32x8;
typedef int i32x4 __attribute__((vector_size(16)));

// ---------------- fragment loader ----------------
// 16-bit A-matrix 16x32 layout (ISA 7.12.2), wave32:
//   lane L: row = L%16, half = L/16
//   elems 0..7  = K[ 8*half .. 8*half+7 ],  elems 8..15 = K[ 16+8*half .. ]
// B (32x16) mirrors A indexed by N, so W^T tiles are loaded as W rows.
__device__ __forceinline__ bf16x16 load_frag(const __bf16* p) {
    bf16x8 lo = *(const bf16x8*)(p);
    bf16x8 hi = *(const bf16x8*)(p + 16);
    return __builtin_shufflevector(lo, hi, 0,1,2,3,4,5,6,7,8,9,10,11,12,13,14,15);
}

// 16-byte global -> LDS stage, async if the toolchain has the builtin
__device__ __forceinline__ void stage16(const __bf16* gsrc, __bf16* ldst) {
#if __has_builtin(__builtin_amdgcn_global_load_async_to_lds_b128)
    __builtin_amdgcn_global_load_async_to_lds_b128(
        (GLOBAL_AS i32x4*)gsrc, (LOCAL_AS i32x4*)ldst, 0, 0);
#else
    *(LOCAL_AS bf16x8*)ldst = *(const GLOBAL_AS bf16x8*)gsrc;
#endif
}

__device__ __forceinline__ void wait_async_then_bar() {
#if __has_builtin(__builtin_amdgcn_global_load_async_to_lds_b128)
#if __has_builtin(__builtin_amdgcn_s_wait_asynccnt)
    __builtin_amdgcn_s_wait_asynccnt(0);
#else
    asm volatile("s_wait_asynccnt 0x0" ::: "memory");
#endif
#endif
    __syncthreads();
}

// ---------------- deterministic gaussian noise (all 32-bit) ----------------
__device__ __forceinline__ float gnoise(uint32_t t, uint32_t idx) {
    uint32_t s = idx * 0x9E3779B9u + t * 0x85EBCA6Bu;
    s ^= s >> 16; s *= 0x7FEB352Du;
    s ^= s >> 15; s *= 0x846CA68Bu;
    s ^= s >> 16;
    uint32_t s2 = s * 0x2C1B3C6Du + 0x297A2D39u;
    s2 ^= s2 >> 15; s2 *= 0x9E3779B9u; s2 ^= s2 >> 16;
    float u1 = ((float)s + 1.0f) * 2.3283064365386963e-10f;   // (0,1]
    float u2 = ((float)s2) * 2.3283064365386963e-10f;         // [0,1)
    return __fsqrt_rn(-2.0f * __logf(u1)) * __cosf(6.283185307179586f * u2);
}

// ---------------- device-wide barrier ----------------
__device__ __forceinline__ void grid_barrier(unsigned* cnt, unsigned* gen) {
    asm volatile("s_wait_storecnt 0x0" ::: "memory");  // activations visible in L2
    __syncthreads();
    if (threadIdx.x == 0) {
        __threadfence();
        unsigned g = __hip_atomic_load(gen, __ATOMIC_RELAXED, __HIP_MEMORY_SCOPE_AGENT);
        unsigned a = __hip_atomic_fetch_add(cnt, 1u, __ATOMIC_ACQ_REL, __HIP_MEMORY_SCOPE_AGENT);
        if (a == NBLOCKS - 1u) {
            __hip_atomic_store(cnt, 0u, __ATOMIC_RELAXED, __HIP_MEMORY_SCOPE_AGENT);
            __hip_atomic_fetch_add(gen, 1u, __ATOMIC_RELEASE, __HIP_MEMORY_SCOPE_AGENT);
        } else {
            while (__hip_atomic_load(gen, __ATOMIC_ACQUIRE, __HIP_MEMORY_SCOPE_AGENT) == g) {
                __builtin_amdgcn_s_sleep(2);
            }
        }
    }
    __syncthreads();
}

// ---------------- prep kernels ----------------
__global__ void prep_wrec(const float* __restrict__ W, __bf16* __restrict__ Wb) {
    int i = blockIdx.x * 256 + threadIdx.x;
    if (i >= H_SZ * H_SZ) return;
    int h = i / H_SZ, k = i - h * H_SZ;
    float m = 1.0f;
    if (h <  H1_SZ  && k >= H12_SZ) m = 0.0f;   // h1 rows: no read of h3
    if (h >= H12_SZ && k <  H1_SZ ) m = 0.0f;   // h3 rows: no read of h1
    Wb[i] = (__bf16)(W[i] * m);
}

__global__ void prep_win(const float* __restrict__ W, __bf16* __restrict__ Wb) {
    int i = blockIdx.x * 256 + threadIdx.x;
    if (i >= H_SZ * IN_SZ) return;
    int h = i / IN_SZ, c = i - h * IN_SZ;
    float m;
    if      (h < H1_SZ)  m = (c < IN_HT) ? 1.0f : 0.0f;                 // heading+targcolor
    else if (h < H12_SZ) m = (c >= IN_HT) ? 1.0f : 0.0f;                // rules only
    else                 m = (c >= IN_Hh && c < IN_HT) ? 1.0f : 0.0f;   // targcolor only
    Wb[i] = (__bf16)(W[i] * m);
}

__global__ void prep_x(const float* __restrict__ X, __bf16* __restrict__ Xb, int n) {
    int i = blockIdx.x * 256 + threadIdx.x;
    if (i < n) Xb[i] = (__bf16)X[i];
}

__global__ void prep_zero(__bf16* __restrict__ buf0, unsigned* __restrict__ bar) {
    int i = blockIdx.x * 256 + threadIdx.x;
    if (i < B_SZ * H_SZ) buf0[i] = (__bf16)0.0f;
    if (i < 2) bar[i] = 0u;
}

// ---------------- persistent RNN kernel ----------------
// 112 blocks x 128 threads (4 waves). Wave w owns a 16(M) x 64(N) patch
// (4 WMMA accumulator chains) for the whole run; hidden state lives in
// registers; ReLU activations ping-pong through bf16 global buffers. Each
// step: async-stage the shared 16-row activation tile into LDS, overlap the
// input-projection GEMM (x fragments straight from L2) with the stage, then
// run the big recurrent GEMM from LDS.
__global__ void __launch_bounds__(THREADS)
rnn_persistent(const __bf16* __restrict__ Wrec,   // [H,H]  masked bf16
               const __bf16* __restrict__ Win,    // [H,IN] masked bf16
               const __bf16* __restrict__ Xbf,    // [T,B,IN] bf16
               const float*  __restrict__ b_in,   // [H]
               const float*  __restrict__ b_rec,  // [H]
               __bf16* buf0, __bf16* buf1,        // [B,H] ping-pong activations
               float* __restrict__ outAll,        // [T,B,H] ++ state[B,H] ++ out[B,H]
               unsigned* cnt, unsigned* gen) {
    __shared__ alignas(16) __bf16 ldsA[16 * LDA];   // 57,600 B

    const int tid   = threadIdx.x;
    const int wave  = tid >> 5;
    const int lane  = tid & 31;
    const int half  = lane >> 4;
    const int l16   = lane & 15;
    const int tileN = blockIdx.x % NTILE_BLK;
    const int tileM = blockIdx.x / NTILE_BLK;
    const int mBase = tileM * 16;
    const int n0    = tileN * 256 + wave * 64;    // 4 N tiles: n0 + 16*j

    // B-fragment row pointers (Wrec rows == W^T columns), plus Win rows
    const __bf16* bRow[4];
    const __bf16* wiRow[4];
    float bias[4], sig[4];
    int   nc[4];
    #pragma unroll
    for (int j = 0; j < 4; ++j) {
        const int r = n0 + 16 * j + l16;
        bRow[j]  = Wrec + (size_t)r * H_SZ  + 8 * half;
        wiRow[j] = Win  + (size_t)r * IN_SZ + 8 * half;
        nc[j]    = r;
        bias[j]  = b_in[r] + b_rec[r];
        sig[j]   = (r >= H2_SZ && r < H12_SZ) ? 0.0f : SIG_F;   // faithful sigma bug
    }

    float st[4][8];
    #pragma unroll
    for (int j = 0; j < 4; ++j)
        #pragma unroll
        for (int i = 0; i < 8; ++i) st[j][i] = 0.0f;

    __bf16* bufs[2] = {buf0, buf1};
    const size_t TBH = (size_t)T_STEPS * B_SZ * H_SZ;
    const size_t BH  = (size_t)B_SZ * H_SZ;

    for (int t = 0; t < T_STEPS; ++t) {
        const __bf16* Aprev = bufs[t & 1];
        __bf16*       Anext = bufs[(t + 1) & 1];

        // ---- issue async stage of A tile (16 x H) into LDS ----
        {
            const __bf16* aSrc = Aprev + (size_t)mBase * H_SZ;
            #pragma unroll 4
            for (int c = tid; c < 16 * (H_SZ / 8); c += THREADS) {
                const int r = c / (H_SZ / 8), col8 = c - r * (H_SZ / 8);
                stage16(aSrc + (size_t)r * H_SZ + col8 * 8, &ldsA[r * LDA + col8 * 8]);
            }
        }

        f32x8 acc[4];
        #pragma unroll
        for (int j = 0; j < 4; ++j) acc[j] = {};

        // ---- input projection (overlaps async stage): x_t[16,IN] x Win^T[IN,64]
        //      x fragments read directly from global (tiny, L2-resident)
        const __bf16* xRow = Xbf + ((size_t)t * B_SZ + mBase + l16) * IN_SZ + 8 * half;
        #pragma unroll
        for (int k = 0; k < IN_SZ; k += 32) {
            bf16x16 a = load_frag(xRow + k);
            #pragma unroll
            for (int j = 0; j < 4; ++j) {
                bf16x16 b = load_frag(wiRow[j] + k);
                acc[j] = __builtin_amdgcn_wmma_f32_16x16x32_bf16(
                            false, a, false, b, (short)0, acc[j], false, false);
            }
        }

        // ---- wait for staged tile, then recurrent GEMM from LDS ----
        wait_async_then_bar();
        const __bf16* aFragBase = &ldsA[l16 * LDA + 8 * half];
        #pragma unroll 2
        for (int k = 0; k < H_SZ; k += 32) {
            __builtin_prefetch(bRow[0] + k + 64, 0, 3);   // unconditional, branch-free
            bf16x16 a = load_frag(aFragBase + k);
            #pragma unroll
            for (int j = 0; j < 4; ++j) {
                bf16x16 b = load_frag(bRow[j] + k);
                acc[j] = __builtin_amdgcn_wmma_f32_16x16x32_bf16(
                            false, a, false, b, (short)0, acc[j], false, false);
            }
        }

        // ---- elementwise step + outputs ----
        float* outT = outAll + (size_t)t * BH;
        #pragma unroll
        for (int i = 0; i < 8; ++i) {
            const int    row  = mBase + 8 * half + i;   // C/D: rows 8*half + i
            const size_t base = (size_t)row * H_SZ;
            #pragma unroll
            for (int j = 0; j < 4; ++j) {
                float total = acc[j][i] + bias[j];
                float s = st[j][i] * (1.0f - ALPHA_F) + total * ALPHA_F;
                if (sig[j] != 0.0f)
                    s += sig[j] * gnoise((uint32_t)t, (uint32_t)(base + nc[j]));
                st[j][i] = s;
                float o = fmaxf(s, 0.0f);
                outT[base + nc[j]] = o;
                *(GLOBAL_AS __bf16*)(Anext + base + nc[j]) = (__bf16)o;
                if (t == T_STEPS - 1) {
                    outAll[TBH + base + nc[j]]      = s;   // final state
                    outAll[TBH + BH + base + nc[j]] = o;   // final out
                }
            }
        }
        grid_barrier(cnt, gen);
    }
}

// ---------------- host launcher ----------------
extern "C" void kernel_launch(void* const* d_in, const int* in_sizes, int n_in,
                              void* d_out, int out_size, void* d_ws, size_t ws_size,
                              hipStream_t stream) {
    const float* inputs = (const float*)d_in[0];   // [T,B,IN]
    const float* W_in   = (const float*)d_in[1];   // [H,IN]
    const float* b_in   = (const float*)d_in[2];   // [H]
    const float* W_rec  = (const float*)d_in[3];   // [H,H]
    const float* b_rec  = (const float*)d_in[4];   // [H]
    (void)in_sizes; (void)n_in; (void)out_size; (void)ws_size;

    char* ws = (char*)d_ws;
    // workspace layout (all 256B aligned)
    __bf16*   Wrecb = (__bf16*)(ws + 0);            //  6,422,528 B
    __bf16*   Winb  = (__bf16*)(ws + 6422528);      //    573,440 B
    __bf16*   Xbf   = (__bf16*)(ws + 6995968);      // 20,971,520 B
    __bf16*   buf0  = (__bf16*)(ws + 27967488);     //    917,504 B
    __bf16*   buf1  = (__bf16*)(ws + 28884992);     //    917,504 B
    unsigned* bar   = (unsigned*)(ws + 29802496);   //  cnt, gen

    const int nWrec = H_SZ * H_SZ;
    const int nWin  = H_SZ * IN_SZ;
    const int nX    = T_STEPS * B_SZ * IN_SZ;
    const int nBuf  = B_SZ * H_SZ;

    prep_wrec<<<(nWrec + 255) / 256, 256, 0, stream>>>(W_rec, Wrecb);
    prep_win <<<(nWin  + 255) / 256, 256, 0, stream>>>(W_in, Winb);
    prep_x   <<<(nX    + 255) / 256, 256, 0, stream>>>(inputs, Xbf, nX);
    prep_zero<<<(nBuf  + 255) / 256, 256, 0, stream>>>(buf0, bar);

    rnn_persistent<<<NBLOCKS, THREADS, 0, stream>>>(
        Wrecb, Winb, Xbf, b_in, b_rec, buf0, buf1,
        (float*)d_out, bar + 0, bar + 1);
}
